// NoteUnrollingDecoder_19713899889460
// MI455X (gfx1250) — matrix operand order repaired and checked by reading, not compiled, not run on hardware
//
#include <hip/hip_runtime.h>
#include <hip/hip_bf16.h>

// ---------------------------------------------------------------------------
// NoteUnrollingDecoder for MI455X (gfx1250, wave32, WMMA bf16)
// Persistent per-batch-tile workgroups: 8 blocks x 512 threads run the whole
// L=100 step recurrence with only intra-block barriers. Weights pre-packed to
// bf16 WMMA-fragment-major layout in d_ws (L2-resident, ~9MB).
// ---------------------------------------------------------------------------

#ifdef __HIP_DEVICE_COMPILE__
typedef __bf16 bf16_t;
#else
typedef unsigned short bf16_t;   // same size/alignment for host sizeof()
#endif

typedef __attribute__((ext_vector_type(16))) __bf16 v16bf;
typedef __attribute__((ext_vector_type(8)))  __bf16 v8bf;
typedef __attribute__((ext_vector_type(8)))  float  v8f;

#define NB   128   // batch
#define NH   256   // hidden
#define NV   130   // vocab
#define NVP  144   // vocab padded to 16
#define TB   16    // batch tile per block
#define NTHR 512   // threads per block (16 wave32)
#define NWAV 16

// per-stream packed weight sizes (bf16 elements)
#define SZ_WIH 393216u   // K=512, Npad=768 -> 48*16*512
#define SZ_WHH 196608u   // K=256, Npad=768 -> 48*8*512
#define SZ_OUT 36864u    // K=256, Npad=144 -> 9*8*512
#define SZ_CWI 589824u   // K=768, Npad=768 -> 48*24*512
#define SZ_HID 65536u    // K=256, Npad=256 -> 16*8*512

struct __align__(16) Smem {
  bf16_t z[TB * NH];            // z tile, bf16
  bf16_t note[3 * TB * NH];     // current note embeddings
  bf16_t g1h[3 * TB * NH];
  bf16_t g2h[3 * TB * NH];
  bf16_t unr[3 * TB * NH];      // "unroll"
  bf16_t ctx[TB * NH];
  bf16_t tmp[TB * NH];          // temp for in-place GRU updates
  bf16_t sum[TB * NH];          // unroll[i] + h2 (logits input)
  float  logits[TB * NVP];
  int    amax[TB];
};

#ifdef __HIP_DEVICE_COMPILE__

union Frag { v16bf v; v8bf h[2]; };

__device__ __forceinline__ v8bf ld8(const __bf16* p) {
  return *(const v8bf*)p;
}

// A fragment (16x32 bf16, MxK): row = M (= lane&15), elements in the packed-B
// layout: K = 16*half + e  -> two contiguous 16B chunks.
__device__ __forceinline__ v16bf afrag(const __bf16* row, int half) {
  Frag f;
  f.h[0] = ld8(row + 16 * half);
  f.h[1] = ld8(row + 16 * half + 8);
  return f.v;
}

// B fragment from packed weights: 32 contiguous bytes per lane.
__device__ __forceinline__ v16bf bfrag(const __bf16* p) {
  Frag f;
  f.h[0] = ld8(p);
  f.h[1] = ld8(p + 8);
  return f.v;
}

__device__ __forceinline__ v8f wmma_bf16(v16bf a, v16bf b, v8f c) {
  return __builtin_amdgcn_wmma_f32_16x16x32_bf16(
      false, a, false, b, (short)0, c, false, false);
}

__device__ __forceinline__ float sigm(float x) {
  return 1.0f / (1.0f + __expf(-x));
}

// One wave computes a 16-wide hidden-column tile of one GRU cell update:
//   h' = (1-u)*n + u*h ;  r,u = sigmoid(gi+gh) ; n = tanh(inn + r*hn)
// x = concat of np pieces (each [16][256] bf16, row stride 256).
// Gate accumulators kept separate so everything fuses in registers.
__device__ void gru_unit(const __bf16* x0, const __bf16* x1, const __bf16* x2,
                         int np, const __bf16* h,
                         const __bf16* wihP, const __bf16* whhP,
                         const float* bih, const float* bhh,
                         int tile, __bf16* dst, int lane) {
  const int li = lane & 15, half = lane >> 4;
  v8f Ar = {}, Az = {}, Ani = {}, Anh = {};
  const int KTi = np * 8;                       // k-tiles of 32 over x
  // gi = x @ wih.T   (N=768 gate space: r at tile, z at tile+16, n at tile+32)
  for (int kt = 0; kt < KTi; ++kt) {
    const __bf16* xp = (kt < 8) ? x0 : ((kt < 16) ? x1 : x2);
    const int kb = (kt & 7) * 32;
    v16bf a = afrag(xp + li * NH + kb, half);
    const __bf16* wb = wihP + (size_t)(tile * KTi + kt) * 512 + lane * 16;
    __builtin_prefetch(wb + 512, 0, 0);         // global_prefetch next block
    Ar  = wmma_bf16(a, bfrag(wb), Ar);
    Az  = wmma_bf16(a, bfrag(wb + (size_t)16 * KTi * 512), Az);
    Ani = wmma_bf16(a, bfrag(wb + (size_t)32 * KTi * 512), Ani);
  }
  // gh = h @ whh.T   (K=256)
  for (int kt = 0; kt < 8; ++kt) {
    v16bf a = afrag(h + li * NH + kt * 32, half);
    const __bf16* wb = whhP + (size_t)(tile * 8 + kt) * 512 + lane * 16;
    Ar  = wmma_bf16(a, bfrag(wb), Ar);
    Az  = wmma_bf16(a, bfrag(wb + (size_t)16 * 8 * 512), Az);
    Anh = wmma_bf16(a, bfrag(wb + (size_t)32 * 8 * 512), Anh);
  }
  const int col = tile * 16 + li;
  const float b_r = bih[col] + bhh[col];
  const float b_z = bih[col + 256] + bhh[col + 256];
  const float bi_n = bih[col + 512], bh_n = bhh[col + 512];
  for (int e = 0; e < 8; ++e) {                 // D layout: M = e + 8*half
    const int m = e + 8 * half;
    float r  = sigm(Ar[e] + b_r);
    float u  = sigm(Az[e] + b_z);
    float nn = tanhf(Ani[e] + bi_n + r * (Anh[e] + bh_n));
    float hv = (float)h[m * NH + col];
    dst[m * NH + col] = (__bf16)((1.0f - u) * nn + u * hv);
  }
}

#endif  // __HIP_DEVICE_COMPILE__

// ---------------------------------------------------------------------------
// Prep: fp32 W[Nreal][K] (x@W.T semantics) -> bf16 fragment-major blocks.
// Block (kt,nt) holds 32x16; lane L owns 16 contiguous bf16:
//   dst[(nt*kTiles+kt)*512 + L*16 + e] = W[nt*16+(L&15)][kt*32+16*(L>>4)+e]
// ---------------------------------------------------------------------------
__global__ void pack_kernel(const float* __restrict__ W, void* dstv,
                            int K, int Nreal, int kTiles, long long total) {
#ifdef __HIP_DEVICE_COMPILE__
  __bf16* dst = (__bf16*)dstv;
  for (long long p = blockIdx.x * (long long)blockDim.x + threadIdx.x;
       p < total; p += (long long)gridDim.x * blockDim.x) {
    const int e = (int)(p & 15);
    const int L = (int)((p >> 4) & 31);
    const long long blk = p >> 9;
    const int kt = (int)(blk % kTiles);
    const int nt = (int)(blk / kTiles);
    const int n = nt * 16 + (L & 15);
    const int k = kt * 32 + 16 * (L >> 4) + e;
    const float v = (n < Nreal) ? W[(size_t)n * K + k] : 0.0f;
    dst[p] = (__bf16)v;
  }
#endif
}

// ---------------------------------------------------------------------------
// Persistent decoder: blockIdx.x = batch tile (0..7), 512 threads = 16 waves.
// ---------------------------------------------------------------------------
__global__ void __launch_bounds__(NTHR)
decode_kernel(const float* __restrict__ zg,
              const float* __restrict__ g1bih, const float* __restrict__ g1bhh,
              const float* __restrict__ g2bih, const float* __restrict__ g2bhh,
              const float* __restrict__ cbih,  const float* __restrict__ cbhh,
              const float* __restrict__ outb,  const float* __restrict__ hidb,
              const float* __restrict__ emb,   const int* __restrict__ lenp,
              const void* g1wihPv, const void* g1whhPv,
              const void* g2wihPv, const void* g2whhPv,
              const void* cwihPv,  const void* cwhhPv,
              const void* outPv,   const void* hidPv,
              float* __restrict__ out) {
#ifdef __HIP_DEVICE_COMPILE__
  extern __shared__ char smem_raw[];
  Smem* S = (Smem*)smem_raw;
  const __bf16* g1wihP = (const __bf16*)g1wihPv;
  const __bf16* g1whhP = (const __bf16*)g1whhPv;
  const __bf16* g2wihP = (const __bf16*)g2wihPv;
  const __bf16* g2whhP = (const __bf16*)g2whhPv;
  const __bf16* cwihP  = (const __bf16*)cwihPv;
  const __bf16* cwhhP  = (const __bf16*)cwhhPv;
  const __bf16* outP   = (const __bf16*)outPv;
  const __bf16* hidP   = (const __bf16*)hidPv;

  const int tid = threadIdx.x, lane = tid & 31, wave = tid >> 5;
  const int li = lane & 15, half = lane >> 4;
  const int bt = blockIdx.x;
  const int Lval = *lenp;

  // --- load z tile (bf16), init note = emb[:,0] broadcast ---
  for (int i = tid; i < TB * NH; i += NTHR) {
    const int m = i >> 8, c = i & 255;
    S->z[i] = (__bf16)zg[(size_t)(bt * TB + m) * NH + c];
  }
  for (int i = tid; i < 3 * TB * NH; i += NTHR) {
    const int s = i >> 12, c = i & 255;
    S->note[i] = (__bf16)emb[(size_t)(s * NV) * NH + c];
  }
  __syncthreads();

  // --- hidden = tanh(z @ hid_w.T + hid_b); broadcast to g1h,g2h,ctx ---
  {
    const int tile = wave;  // 16 tiles over 16 waves
    v8f acc = {};
    for (int kt = 0; kt < 8; ++kt) {
      v16bf a = afrag((const __bf16*)S->z + li * NH + kt * 32, half);
      acc = wmma_bf16(a, bfrag(hidP + (size_t)(tile * 8 + kt) * 512 + lane * 16), acc);
    }
    const int col = tile * 16 + li;
    const float hb = hidb[col];
    for (int e = 0; e < 8; ++e) {
      const int m = e + 8 * half, o = m * NH + col;
      const __bf16 v = (__bf16)tanhf(acc[e] + hb);
      S->g1h[o] = v; S->g1h[4096 + o] = v; S->g1h[8192 + o] = v;
      S->g2h[o] = v; S->g2h[4096 + o] = v; S->g2h[8192 + o] = v;
      S->ctx[o] = v;
    }
  }
  __syncthreads();

  for (int t = 0; t < Lval; ++t) {
    // phase 1: g1h per stream (write to unr to avoid RAW race, then copy back)
    for (int u = wave; u < 48; u += NWAV) {
      const int s = u >> 4, tile = u & 15;
      gru_unit((const __bf16*)S->note + s * 4096, (const __bf16*)S->z, nullptr, 2,
               (const __bf16*)S->g1h + s * 4096,
               g1wihP + (size_t)s * SZ_WIH, g1whhP + (size_t)s * SZ_WHH,
               g1bih + s * 768, g1bhh + s * 768,
               tile, (__bf16*)S->unr + s * 4096, lane);
    }
    __syncthreads();
    for (int i = tid; i < 3 * TB * NH; i += NTHR) S->g1h[i] = S->unr[i];
    __syncthreads();

    for (int i3 = 0; i3 < 3; ++i3) {
      // (a) context = GRU(concat(unr0,unr1,unr2), ctx)
      gru_unit((const __bf16*)S->unr, (const __bf16*)S->unr + 4096,
               (const __bf16*)S->unr + 8192, 3,
               (const __bf16*)S->ctx, cwihP, cwhhP, cbih, cbhh,
               wave, (__bf16*)S->tmp, lane);
      __syncthreads();
      for (int i = tid; i < TB * NH; i += NTHR) S->ctx[i] = S->tmp[i];
      __syncthreads();

      // (b) h2 = GRU(concat(ctx,z), g2h[i3])
      gru_unit((const __bf16*)S->ctx, (const __bf16*)S->z, nullptr, 2,
               (const __bf16*)S->g2h + i3 * 4096,
               g2wihP + (size_t)i3 * SZ_WIH, g2whhP + (size_t)i3 * SZ_WHH,
               g2bih + i3 * 768, g2bhh + i3 * 768,
               wave, (__bf16*)S->tmp, lane);
      __syncthreads();
      for (int i = tid; i < TB * NH; i += NTHR) {
        const __bf16 h2 = ((__bf16*)S->tmp)[i];
        S->g2h[i3 * 4096 + i] = h2;
        S->sum[i] = (__bf16)((float)((__bf16*)S->unr)[i3 * 4096 + i] + (float)h2);
      }
      __syncthreads();

      // (c) logits = (unr[i3]+h2) @ out_w[i3].T + out_b ; store + keep in LDS
      if (wave < 9) {
        const int nt = wave;
        v8f acc = {};
        for (int kt = 0; kt < 8; ++kt) {
          v16bf a = afrag((const __bf16*)S->sum + li * NH + kt * 32, half);
          acc = wmma_bf16(a, bfrag(outP + (size_t)i3 * SZ_OUT +
                                   (size_t)(nt * 8 + kt) * 512 + lane * 16), acc);
        }
        const int col = nt * 16 + li;
        if (col < NV) {
          const float bb = outb[i3 * NV + col];
          for (int e = 0; e < 8; ++e) {
            const int m = e + 8 * half;
            const float v = acc[e] + bb;
            S->logits[m * NVP + col] = v;
            const int gb = bt * TB + m;
            out[(((size_t)i3 * NB + gb) * Lval + t) * NV + col] = v;
          }
        }
      }
      __syncthreads();

      // (d) greedy argmax (first-max tie-break, matches jnp.argmax) + embed
      if (tid < TB) {
        const float* row = S->logits + tid * NVP;
        float best = row[0]; int bi = 0;
        for (int v = 1; v < NV; ++v) {
          const float x = row[v];
          if (x > best) { best = x; bi = v; }
        }
        S->amax[tid] = bi;
      }
      __syncthreads();
      for (int i = tid; i < TB * NH; i += NTHR) {
        const int m = i >> 8, c = i & 255;
        S->note[i3 * 4096 + i] =
            (__bf16)emb[((size_t)i3 * NV + S->amax[m]) * NH + c];
      }
      __syncthreads();

      // (e) unr[i3] = GRU(concat(note[i3],z), unr[i3], gru1 stream-1 weights)
      gru_unit((const __bf16*)S->note + i3 * 4096, (const __bf16*)S->z, nullptr, 2,
               (const __bf16*)S->unr + i3 * 4096,
               g1wihP + SZ_WIH, g1whhP + SZ_WHH,
               g1bih + 768, g1bhh + 768,
               wave, (__bf16*)S->tmp, lane);
      __syncthreads();
      for (int i = tid; i < TB * NH; i += NTHR) S->unr[i3 * 4096 + i] = S->tmp[i];
      __syncthreads();
    }
  }
#endif
}

// ---------------------------------------------------------------------------
extern "C" void kernel_launch(void* const* d_in, const int* in_sizes, int n_in,
                              void* d_out, int out_size, void* d_ws, size_t ws_size,
                              hipStream_t stream) {
  const float* z      = (const float*)d_in[0];
  const float* g1wih  = (const float*)d_in[1];
  const float* g1whh  = (const float*)d_in[2];
  const float* g1bih  = (const float*)d_in[3];
  const float* g1bhh  = (const float*)d_in[4];
  const float* g2wih  = (const float*)d_in[5];
  const float* g2whh  = (const float*)d_in[6];
  const float* g2bih  = (const float*)d_in[7];
  const float* g2bhh  = (const float*)d_in[8];
  const float* cwih   = (const float*)d_in[9];
  const float* cwhh   = (const float*)d_in[10];
  const float* cbih   = (const float*)d_in[11];
  const float* cbhh   = (const float*)d_in[12];
  const float* outw   = (const float*)d_in[13];
  const float* outb   = (const float*)d_in[14];
  const float* hidw   = (const float*)d_in[15];
  const float* hidb   = (const float*)d_in[16];
  const float* emb    = (const float*)d_in[17];
  const int*   lenp   = (const int*)d_in[18];

  // workspace layout (bytes); all packed bf16
  char* ws = (char*)d_ws;
  void* g1wihP = ws;                          // 3 * 393216 * 2
  void* g1whhP = ws + 2359296;                // 3 * 196608 * 2
  void* g2wihP = ws + 3538944;
  void* g2whhP = ws + 5898240;
  void* cwihP  = ws + 7077888;                // 589824 * 2
  void* cwhhP  = ws + 8257536;                // 196608 * 2
  void* outP   = ws + 8650752;                // 3 * 36864 * 2
  void* hidP   = ws + 8871936;                // 65536 * 2  -> end 9003008

  auto pack = [&](const float* W, void* dst, int K, int Nreal, int kTiles,
                  long long total) {
    int grid = (int)((total + 255) / 256);
    if (grid > 4096) grid = 4096;
    hipLaunchKernelGGL(pack_kernel, dim3(grid), dim3(256), 0, stream,
                       W, dst, K, Nreal, kTiles, total);
  };

  for (int s = 0; s < 3; ++s) {
    pack(g1wih + (size_t)s * 768 * 512, (char*)g1wihP + (size_t)s * SZ_WIH * 2,
         512, 768, 16, SZ_WIH);
    pack(g1whh + (size_t)s * 768 * 256, (char*)g1whhP + (size_t)s * SZ_WHH * 2,
         256, 768, 8, SZ_WHH);
    pack(g2wih + (size_t)s * 768 * 512, (char*)g2wihP + (size_t)s * SZ_WIH * 2,
         512, 768, 16, SZ_WIH);
    pack(g2whh + (size_t)s * 768 * 256, (char*)g2whhP + (size_t)s * SZ_WHH * 2,
         256, 768, 8, SZ_WHH);
    pack(outw + (size_t)s * 130 * 256, (char*)outP + (size_t)s * SZ_OUT * 2,
         256, 130, 8, SZ_OUT);
  }
  pack(cwih, cwihP, 768, 768, 24, SZ_CWI);
  pack(cwhh, cwhhP, 256, 768, 8, SZ_WHH);
  pack(hidw, hidP, 256, 256, 8, SZ_HID);

  const size_t smem = sizeof(Smem);
  (void)hipFuncSetAttribute((const void*)decode_kernel,
                            hipFuncAttributeMaxDynamicSharedMemorySize,
                            (int)smem);
  hipLaunchKernelGGL(decode_kernel, dim3(NB / TB), dim3(NTHR), smem, stream,
                     z, g1bih, g1bhh, g2bih, g2bhh, cbih, cbhh, outb, hidb,
                     emb, lenp,
                     g1wihP, g1whhP, g2wihP, g2whhP, cwihP, cwhhP, outP, hidP,
                     (float*)d_out);
  (void)in_sizes; (void)n_in; (void)out_size; (void)ws_size;
}
